// TransformerLanguageModel_86878598464157
// MI455X (gfx1250) — matrix operand order repaired and checked by reading, not compiled, not run on hardware
//
#include <hip/hip_runtime.h>
#include <cstdint>

// ---------------- model constants ----------------
constexpr int cB  = 4;
constexpr int cT  = 1024;
constexpr int cC  = 1024;
constexpr int cH  = 4;
constexpr int cHS = 256;
constexpr int cL  = 4;
constexpr int cF  = 4096;     // 4*C
constexpr int cV  = 32000;

// ---------------- vector types -------------------
typedef __attribute__((ext_vector_type(16))) __bf16  v16bf;
typedef __attribute__((ext_vector_type(8)))  float   v8f;
typedef __attribute__((ext_vector_type(4)))  unsigned int u32x4;

union FragBF {
    v16bf v;
    u32x4 q[2];
};

// fp32 -> bf16 round-to-nearest-even
__device__ __forceinline__ unsigned short f2bf(float f) {
    unsigned u = __float_as_uint(f);
    unsigned r = u + 0x7FFFu + ((u >> 16) & 1u);
    return (unsigned short)(r >> 16);
}

// CDNA5 async global->LDS copy (ASYNCcnt-tracked), per CDNA5 ISA
__device__ __forceinline__ void async_b128(unsigned lds_byte_addr, const void* gaddr) {
    asm volatile("global_load_async_to_lds_b128 %0, %1, off"
                 :: "v"(lds_byte_addr), "v"(gaddr) : "memory");
}
#define WAIT_ASYNC_LE4() asm volatile("s_wait_asynccnt 0x4" ::: "memory")
#define WAIT_ASYNC_0()   asm volatile("s_wait_asynccnt 0x0" ::: "memory")

// ---------------- fp32 -> bf16 transpose-convert (R x Cc -> Cc x R), per-z matrix ----------------
__global__ void cvt_T_kernel(const float* __restrict__ in,
                             unsigned short* __restrict__ out,
                             int R, int Cc) {
    __shared__ float tile[32][33];
    long long z = blockIdx.z;
    const float* src     = in  + z * (long long)R * Cc;
    unsigned short* dst  = out + z * (long long)R * Cc;
    int r0 = blockIdx.y * 32, c0 = blockIdx.x * 32;
    int tx = threadIdx.x, ty = threadIdx.y;              // 32 x 8
    #pragma unroll
    for (int i = 0; i < 4; ++i)
        tile[ty + i * 8][tx] = src[(long long)(r0 + ty + i * 8) * Cc + c0 + tx];
    __syncthreads();
    #pragma unroll
    for (int i = 0; i < 4; ++i)
        dst[(long long)(c0 + ty + i * 8) * R + r0 + tx] = f2bf(tile[tx][ty + i * 8]);
}

// ---------------- embedding: x = tok_emb[idx] + pos_emb ----------------
__global__ void embed_kernel(const int* __restrict__ idx,
                             const float* __restrict__ tok,
                             const float* __restrict__ pos,
                             float* __restrict__ x) {
    int row = blockIdx.x;                 // b*T + t
    int t   = row % cT;
    int tk  = idx[row];
    const float* te = tok + (long long)tk * cC;
    const float* pe = pos + (long long)t * cC;
    float* o        = x   + (long long)row * cC;
    for (int i = threadIdx.x; i < cC; i += 256)
        o[i] = te[i] + pe[i];
}

// ---------------- LayerNorm over C=1024, bf16 output ----------------
__global__ void ln_kernel(const float* __restrict__ x,
                          const float* __restrict__ g,
                          const float* __restrict__ bta,
                          unsigned short* __restrict__ out) {
    __shared__ float red[256];
    int row = blockIdx.x, tid = threadIdx.x;
    const float* xr = x + (long long)row * cC;

    float v[4]; float s = 0.f;
    #pragma unroll
    for (int i = 0; i < 4; ++i) { v[i] = xr[tid + i * 256]; s += v[i]; }

    red[tid] = s; __syncthreads();
    #pragma unroll
    for (int off = 128; off > 0; off >>= 1) {
        if (tid < off) red[tid] += red[tid + off];
        __syncthreads();
    }
    float mu = red[0] * (1.0f / cC);
    __syncthreads();

    float s2 = 0.f;
    #pragma unroll
    for (int i = 0; i < 4; ++i) { float d = v[i] - mu; s2 += d * d; }
    red[tid] = s2; __syncthreads();
    #pragma unroll
    for (int off = 128; off > 0; off >>= 1) {
        if (tid < off) red[tid] += red[tid + off];
        __syncthreads();
    }
    float inv = rsqrtf(red[0] * (1.0f / cC) + 1e-5f);

    unsigned short* o = out + (long long)row * cC;
    #pragma unroll
    for (int i = 0; i < 4; ++i) {
        int c = tid + i * 256;
        o[c] = f2bf((v[i] - mu) * inv * g[c] + bta[c]);
    }
}

// ---------------- column softmax (axis = query t), causal, bf16 alpha ----------------
// reference: scores = where(tril, qk, -inf) * (1/sqrt(C)); softmax over axis=2 (t)
__global__ void softmax_cols(const float* __restrict__ scores,
                             unsigned short* __restrict__ alpha) {
    const float scale = 0.03125f;          // 1/sqrt(1024)
    int z = blockIdx.y;                    // (b*H + h)
    int s = blockIdx.x * 256 + threadIdx.x;
    const float* col = scores + (long long)z * cT * cT + s;

    float m = -INFINITY, d = 0.f;
    for (int t = s; t < cT; ++t) {         // t < s is masked (-inf -> 0)
        float v  = col[(long long)t * cT] * scale;
        float mn = fmaxf(m, v);
        d = d * __expf(m - mn) + __expf(v - mn);
        m = mn;
    }
    float inv = 1.0f / d;

    unsigned short* oc = alpha + (long long)z * cT * cT + s;
    for (int t = 0; t < cT; ++t) {
        float a = (t >= s) ? __expf(col[(long long)t * cT] * scale - m) * inv : 0.f;
        oc[(long long)t * cT] = f2bf(a);
    }
}

// ---------------- WMMA GEMM: D = A(bf16, MxK) x B(bf16, NxK "BT") [+bias][+res][relu] ----
// Tiles: BM=BN=128, BK=32. 256 threads = 8 waves (wave32), each wave 32x64.
// Double-buffered LDS fed by global_load_async_to_lds_b128 (ASYNCcnt).
constexpr int GBM = 128, GBN = 128, GBK = 32, GSTR = 40; // padded rows: b128 reads hit all 64 banks

struct GemmP {
    const unsigned short* A;
    const unsigned short* Bm;   // N x K row-major
    const float* bias;
    const float* res;
    void* D;
    int lda, ldb, ldd, ldres, K, inner;
    long long sAo, sAi, sBo, sBi, sDo, sDi;
};

template <bool BIAS, bool RELU, bool RES, bool OBF, bool DT>
__global__ __launch_bounds__(256) void gemm_wmma(GemmP p) {
    __shared__ unsigned short As[2][GBM * GSTR];
    __shared__ unsigned short Bs[2][GBN * GSTR];

    const int tid = threadIdx.x;
    const int z   = blockIdx.z;
    const long long zo = z / p.inner, zi = z % p.inner;
    const unsigned short* A  = p.A  + zo * p.sAo + zi * p.sAi;
    const unsigned short* Bg = p.Bm + zo * p.sBo + zi * p.sBi;

    const int m0 = blockIdx.y * GBM, n0 = blockIdx.x * GBN;

    const int wave = tid >> 5, lane = tid & 31;
    const int l15 = lane & 15, hi = lane >> 4;
    const int wm = (wave >> 1) * 32;   // 4 waves along M
    const int wn = (wave & 1) * 64;    // 2 waves along N

    v8f acc[2][4];
    #pragma unroll
    for (int i = 0; i < 2; ++i)
        #pragma unroll
        for (int j = 0; j < 4; ++j)
            #pragma unroll
            for (int e = 0; e < 8; ++e) acc[i][j][e] = 0.f;

    // tile copy role: 2 threads per row, 32B (16 bf16) each
    const int r  = tid >> 1, cseg = (tid & 1) * 16;
    const unsigned short* gA = A  + (long long)(m0 + r) * p.lda + cseg;
    const unsigned short* gB = Bg + (long long)(n0 + r) * p.ldb + cseg;
    const unsigned ldsA0 = (unsigned)(uintptr_t)&As[0][r * GSTR + cseg];
    const unsigned ldsA1 = (unsigned)(uintptr_t)&As[1][r * GSTR + cseg];
    const unsigned ldsB0 = (unsigned)(uintptr_t)&Bs[0][r * GSTR + cseg];
    const unsigned ldsB1 = (unsigned)(uintptr_t)&Bs[1][r * GSTR + cseg];

    auto issue = [&](int kt, int buf) {
        const int k0 = kt * GBK;
        unsigned la = buf ? ldsA1 : ldsA0;
        unsigned lb = buf ? ldsB1 : ldsB0;
        async_b128(la,      gA + k0);
        async_b128(la + 16, gA + k0 + 8);
        async_b128(lb,      gB + k0);
        async_b128(lb + 16, gB + k0 + 8);
    };

    const int nk = p.K / GBK;
    issue(0, 0);
    for (int kt = 0; kt < nk; ++kt) {
        const int cur = kt & 1;
        if (kt + 1 < nk) { issue(kt + 1, cur ^ 1); WAIT_ASYNC_LE4(); }
        else             { WAIT_ASYNC_0(); }
        __syncthreads();

        FragBF a[2], b[4];
        #pragma unroll
        for (int i = 0; i < 2; ++i) {
            // 16-bit A 16x32 layout: low lanes K{0-7,16-23}, high lanes K{8-15,24-31}
            const u32x4* pA = (const u32x4*)(&As[cur][(wm + i * 16 + l15) * GSTR]);
            a[i].q[0] = pA[hi];
            a[i].q[1] = pA[2 + hi];
        }
        #pragma unroll
        for (int j = 0; j < 4; ++j) {
            // B 32x16: low lanes K{0-7,16-23}, high lanes K{8-15,24-31} of its column
            const u32x4* pB = (const u32x4*)(&Bs[cur][(wn + j * 16 + l15) * GSTR]);
            b[j].q[0] = pB[hi];
            b[j].q[1] = pB[2 + hi];
        }
        #pragma unroll
        for (int i = 0; i < 2; ++i)
            #pragma unroll
            for (int j = 0; j < 4; ++j)
                acc[i][j] = __builtin_amdgcn_wmma_f32_16x16x32_bf16(
                    false, a[i].v, false, b[j].v, (short)0, acc[i][j], false, false);
        __syncthreads();
    }

    // epilogue: C/D f32 16x16 layout: lane(l15,hi), VGPR r -> (M = hi*8+r, N = l15)
    const long long zD = zo * p.sDo + zi * p.sDi;
    if (DT) {
        // store D transposed (N x M), 8 contiguous bf16 -> one 16B store
        #pragma unroll
        for (int j = 0; j < 4; ++j) {
            const int n = n0 + wn + j * 16 + l15;
            #pragma unroll
            for (int i = 0; i < 2; ++i) {
                const int mb = m0 + wm + i * 16 + hi * 8;
                unsigned short tmp[8];
                #pragma unroll
                for (int rr = 0; rr < 8; ++rr) tmp[rr] = f2bf(acc[i][j][rr]);
                *(u32x4*)((unsigned short*)p.D + zD + (long long)n * p.ldd + mb) =
                    *(const u32x4*)tmp;
            }
        }
    } else {
        #pragma unroll
        for (int j = 0; j < 4; ++j) {
            const int n = n0 + wn + j * 16 + l15;
            const float bv = BIAS ? p.bias[n] : 0.f;
            #pragma unroll
            for (int i = 0; i < 2; ++i) {
                #pragma unroll
                for (int rr = 0; rr < 8; ++rr) {
                    const int m = m0 + wm + i * 16 + hi * 8 + rr;
                    float val = acc[i][j][rr] + bv;
                    if (RES)  val += p.res[(long long)m * p.ldres + n];
                    if (RELU) val = fmaxf(val, 0.f);
                    if (OBF)
                        ((unsigned short*)p.D)[zD + (long long)m * p.ldd + n] = f2bf(val);
                    else
                        ((float*)p.D)[zD + (long long)m * p.ldd + n] = val;
                }
            }
        }
    }
}

// ---------------- host orchestration ----------------
extern "C" void kernel_launch(void* const* d_in, const int* in_sizes, int n_in,
                              void* d_out, int out_size, void* d_ws, size_t ws_size,
                              hipStream_t stream) {
    (void)in_sizes; (void)n_in; (void)out_size; (void)ws_size;

    const int*   idx     = (const int*)d_in[0];
    const float* tok_emb = (const float*)d_in[1];
    const float* pos_emb = (const float*)d_in[2];
    const float* wq      = (const float*)d_in[3];
    const float* wk      = (const float*)d_in[4];
    const float* wv      = (const float*)d_in[5];
    const float* proj_w  = (const float*)d_in[6];
    const float* proj_b  = (const float*)d_in[7];
    const float* ln1_g   = (const float*)d_in[8];
    const float* ln1_b   = (const float*)d_in[9];
    const float* ln2_g   = (const float*)d_in[10];
    const float* ln2_b   = (const float*)d_in[11];
    const float* ff_w1   = (const float*)d_in[12];
    const float* ff_b1   = (const float*)d_in[13];
    const float* ff_w2   = (const float*)d_in[14];
    const float* ff_b2   = (const float*)d_in[15];
    const float* lnf_g   = (const float*)d_in[16];
    const float* lnf_b   = (const float*)d_in[17];
    const float* head_w  = (const float*)d_in[18];
    const float* head_b  = (const float*)d_in[19];
    float* out = (float*)d_out;

    // ---- workspace carve-out ----
    char* wsb = (char*)d_ws;
    size_t off = 0;
    auto alloc = [&](size_t bytes) -> void* {
        off = (off + 255) & ~(size_t)255;
        void* p = wsb + off;
        off += bytes;
        return p;
    };
    const long long nW  = (long long)cL * cH * cC * cHS;  // per q/k/v weight
    const long long nP  = (long long)cL * cC * cC;
    const long long nF1 = (long long)cL * cC * cF;
    const long long nHd = (long long)cC * cV;
    const long long nX  = (long long)cB * cT * cC;
    const long long nQ  = (long long)cB * cH * cT * cHS;
    const long long nS  = (long long)cB * cH * cT * cT;
    const long long nHf = (long long)cB * cT * cF;

    // all weights stored TRANSPOSED (N x K) in bf16
    unsigned short* wqT   = (unsigned short*)alloc(nW  * 2);  // (l,h): HS x C
    unsigned short* wkT   = (unsigned short*)alloc(nW  * 2);
    unsigned short* wvT   = (unsigned short*)alloc(nW  * 2);
    unsigned short* projT = (unsigned short*)alloc(nP  * 2);  // l: C x C (transposed)
    unsigned short* ff1T  = (unsigned short*)alloc(nF1 * 2);  // l: F x C
    unsigned short* ff2T  = (unsigned short*)alloc(nF1 * 2);  // l: C x F
    unsigned short* headT = (unsigned short*)alloc(nHd * 2);  // V x C
    float*          x     = (float*)alloc(nX * 4);
    unsigned short* xn_bf = (unsigned short*)alloc(nX * 2);
    unsigned short* q_bf  = (unsigned short*)alloc(nQ * 2);   // (b,h): T x HS
    unsigned short* k_bf  = (unsigned short*)alloc(nQ * 2);   // (b,h): T x HS
    unsigned short* vT_bf = (unsigned short*)alloc(nQ * 2);   // (b,h): HS x T (transposed)
    float*          scores= (float*)alloc(nS * 4);
    unsigned short* alp_bf= (unsigned short*)alloc(nS * 2);
    unsigned short* y_bf  = (unsigned short*)alloc(nX * 2);
    unsigned short* h_bf  = (unsigned short*)alloc(nHf * 2);

    // ---- weight transpose + fp32->bf16 conversion ----
    auto cvtT = [&](const float* src, unsigned short* dst, int R, int Cc, int Z) {
        cvt_T_kernel<<<dim3(Cc / 32, R / 32, Z), dim3(32, 8), 0, stream>>>(src, dst, R, Cc);
    };
    cvtT(wq, wqT, cC, cHS, cL * cH);
    cvtT(wk, wkT, cC, cHS, cL * cH);
    cvtT(wv, wvT, cC, cHS, cL * cH);
    cvtT(proj_w, projT, cC, cC, cL);
    cvtT(ff_w1, ff1T, cC, cF, cL);
    cvtT(ff_w2, ff2T, cF, cC, cL);
    cvtT(head_w, headT, cC, cV, 1);

    // ---- embedding ----
    embed_kernel<<<dim3(cB * cT), dim3(256), 0, stream>>>(idx, tok_emb, pos_emb, x);

    auto gemm_grid = [](int M, int N, int Z) { return dim3(N / GBN, M / GBM, Z); };

    for (int l = 0; l < cL; ++l) {
        // ln1
        ln_kernel<<<dim3(cB * cT), dim3(256), 0, stream>>>(
            x, ln1_g + (long long)l * cC, ln1_b + (long long)l * cC, xn_bf);

        // q/k = xn @ w[l,h]; v additionally stored transposed  (z = b*H + h, inner = H)
        for (int w = 0; w < 3; ++w) {
            const unsigned short* W = (w == 0 ? wqT : w == 1 ? wkT : wvT)
                                      + (long long)l * cH * cC * cHS;
            GemmP p{};
            p.A = xn_bf; p.Bm = W;
            p.lda = cC; p.ldb = cC; p.K = cC; p.inner = cH;
            p.sAo = (long long)cT * cC; p.sAi = 0;
            p.sBo = 0;                  p.sBi = (long long)cC * cHS;
            p.sDo = (long long)cH * cT * cHS; p.sDi = (long long)cT * cHS;
            if (w < 3 - 1) {
                p.D = (w == 0 ? q_bf : k_bf); p.ldd = cHS;
                gemm_wmma<false, false, false, true, false>
                    <<<gemm_grid(cT, cHS, cB * cH), dim3(256), 0, stream>>>(p);
            } else {
                p.D = vT_bf; p.ldd = cT;    // v written as HS x T
                gemm_wmma<false, false, false, true, true>
                    <<<gemm_grid(cT, cHS, cB * cH), dim3(256), 0, stream>>>(p);
            }
        }

        // scores = q @ k^T   (B operand = k as N x K row-major)
        {
            GemmP p{};
            p.A = q_bf; p.Bm = k_bf; p.D = scores;
            p.lda = cHS; p.ldb = cHS; p.ldd = cT; p.K = cHS; p.inner = 1;
            p.sAo = (long long)cT * cHS; p.sAi = 0;
            p.sBo = (long long)cT * cHS; p.sBi = 0;
            p.sDo = (long long)cT * cT;  p.sDi = 0;
            gemm_wmma<false, false, false, false, false>
                <<<gemm_grid(cT, cT, cB * cH), dim3(256), 0, stream>>>(p);
        }

        // causal mask + scale + softmax over query axis -> alpha (bf16)
        softmax_cols<<<dim3(cT / 256, cB * cH), dim3(256), 0, stream>>>(scores, alp_bf);

        // y = alpha @ v  (B operand = v^T as N x K), written into head-concat layout
        {
            GemmP p{};
            p.A = alp_bf; p.Bm = vT_bf; p.D = y_bf;
            p.lda = cT; p.ldb = cT; p.ldd = cC; p.K = cT; p.inner = cH;
            p.sAo = (long long)cH * cT * cT;  p.sAi = (long long)cT * cT;
            p.sBo = (long long)cH * cT * cHS; p.sBi = (long long)cT * cHS;
            p.sDo = (long long)cT * cC;       p.sDi = cHS;
            gemm_wmma<false, false, false, true, false>
                <<<gemm_grid(cT, cHS, cB * cH), dim3(256), 0, stream>>>(p);
        }

        // x = x + y @ proj_w[l] + proj_b[l]
        {
            GemmP p{};
            p.A = y_bf; p.Bm = projT + (long long)l * cC * cC;
            p.bias = proj_b + (long long)l * cC;
            p.res = x; p.ldres = cC; p.D = x;
            p.lda = cC; p.ldb = cC; p.ldd = cC; p.K = cC; p.inner = 1;
            gemm_wmma<true, false, true, false, false>
                <<<gemm_grid(cB * cT, cC, 1), dim3(256), 0, stream>>>(p);
        }

        // ln2
        ln_kernel<<<dim3(cB * cT), dim3(256), 0, stream>>>(
            x, ln2_g + (long long)l * cC, ln2_b + (long long)l * cC, xn_bf);

        // h = relu(xn2 @ ff_w1 + ff_b1)
        {
            GemmP p{};
            p.A = xn_bf; p.Bm = ff1T + (long long)l * cC * cF;
            p.bias = ff_b1 + (long long)l * cF; p.D = h_bf;
            p.lda = cC; p.ldb = cC; p.ldd = cF; p.K = cC; p.inner = 1;
            gemm_wmma<true, true, false, true, false>
                <<<gemm_grid(cB * cT, cF, 1), dim3(256), 0, stream>>>(p);
        }

        // x = x + h @ ff_w2 + ff_b2
        {
            GemmP p{};
            p.A = h_bf; p.Bm = ff2T + (long long)l * cF * cC;
            p.bias = ff_b2 + (long long)l * cC;
            p.res = x; p.ldres = cC; p.D = x;
            p.lda = cF; p.ldb = cF; p.ldd = cC; p.K = cF; p.inner = 1;
            gemm_wmma<true, false, true, false, false>
                <<<gemm_grid(cB * cT, cC, 1), dim3(256), 0, stream>>>(p);
        }
    }

    // final LN
    ln_kernel<<<dim3(cB * cT), dim3(256), 0, stream>>>(x, lnf_g, lnf_b, xn_bf);

    // logits = xnf @ head_w + head_b  (fp32 out to d_out)
    {
        GemmP p{};
        p.A = xn_bf; p.Bm = headT; p.bias = head_b; p.D = out;
        p.lda = cC; p.ldb = cC; p.ldd = cV; p.K = cC; p.inner = 1;
        gemm_wmma<true, false, false, false, false>
            <<<gemm_grid(cB * cT, cV, 1), dim3(256), 0, stream>>>(p);
    }
}
